// DynamicTeacher_4758823764224
// MI455X (gfx1250) — matrix operand order repaired and checked by reading, not compiled, not run on hardware
//
#include <hip/hip_runtime.h>
#include <hip/hip_bf16.h>
#include <math.h>

typedef __attribute__((ext_vector_type(16))) _Float16 v16h;
typedef __attribute__((ext_vector_type(8)))  _Float16 v8h;
typedef __attribute__((ext_vector_type(8)))  float    v8f;

#define WMMA_F16(a, b, c) \
  __builtin_amdgcn_wmma_f32_16x16x32_f16(false, (a), false, (b), (short)0, (c), false, false)

__device__ __forceinline__ int imin(int a, int b) { return a < b ? a : b; }
__device__ __forceinline__ int imax(int a, int b) { return a > b ? a : b; }

// ---------------------------------------------------------------------------
// Fragment loaders matching CDNA5 ISA 7.12.2 (wave32)
// A 16x32 f16: lanes 0-15 hold row M=lane, K in {0..7, 16..23};
//              lanes 16-31 hold same rows, K in {8..15, 24..31}.
// B 32x16 f16 (staged transposed in LDS as Bt[n][k], 32 halfs/row):
//              lanes 0-15 col N=lane, K in {0..7, 16..23}; lanes 16-31 same cols, K in {8..15, 24..31}.
// ---------------------------------------------------------------------------
__device__ __forceinline__ v16h load_a_frag(const _Float16* Asl, int m_sub, int lane) {
  int fr = lane & 15;
  int kh = (lane >> 4) * 8;
  const v8h* p0 = (const v8h*)(Asl + (m_sub + fr) * 32 + kh);
  const v8h* p1 = (const v8h*)(Asl + (m_sub + fr) * 32 + kh + 16);
  v8h lo = *p0, hi = *p1;
  v16h a;
#pragma unroll
  for (int i = 0; i < 8; ++i) { a[i] = lo[i]; a[i + 8] = hi[i]; }
  return a;
}

__device__ __forceinline__ v16h load_b_frag(const _Float16* Bsl, int n_sub, int lane) {
  int fc = lane & 15;
  int kb = (lane >> 4) * 8;
  const v8h* p0 = (const v8h*)(Bsl + (n_sub + fc) * 32 + kb);
  v8h lo = *p0, hi = *(p0 + 2);
  v16h b;
#pragma unroll
  for (int i = 0; i < 8; ++i) { b[i] = lo[i]; b[i + 8] = hi[i]; }
  return b;
}

// ---------------------------------------------------------------------------
// Generic batched GEMM:  D[m,n] = relu?( alpha * (sum_k A(m,k)*B(k,n) + bias[n]) )
// A(m,k) at A + m*a_rs + k*a_cs ; B(k,n) at B + k*b_rs + n*b_cs ;
// D(m,n) at D + m*d_rs + n.   Batched via blockIdx.z strides (64-bit, applied
// once); all per-element offsets are 32-bit (every tensor here is < 2^31 elts).
// Tile: 64(M) x 128(N) x 32(K); 8 waves, each wave = 16x64 (1 A-frag, 4 WMMA).
// LDS double-buffered: one barrier per K-step. Staging order chosen per-operand
// from the strides so the fast-moving lane index follows the contiguous axis.
// ---------------------------------------------------------------------------
__global__ __launch_bounds__(256) void k_gemm(
    const float* __restrict__ A, const float* __restrict__ B,
    const float* __restrict__ bias, float* __restrict__ D,
    int M, int N, int K,
    int a_rs, int a_cs, int b_rs, int b_cs, int d_rs,
    long bsA, long bsB, long bsD, float alpha, int relu)
{
  __shared__ __align__(16) _Float16 Asl[2][64 * 32];
  __shared__ __align__(16) _Float16 Bsl[2][128 * 32];   // Bt[n_local][k_local]

  A += (long)blockIdx.z * bsA;
  B += (long)blockIdx.z * bsB;
  D += (long)blockIdx.z * bsD;

  const int m0 = blockIdx.x * 64;
  const int n0 = blockIdx.y * 128;
  const int tid = threadIdx.x;
  const int lane = tid & 31;
  const int w = tid >> 5;
  const int m_sub = (w & 3) * 16;
  const int n_grp = (w >> 2) * 64;

  // Clamped-index staging: address math is unconditional 32-bit (no exec-mask
  // save/restore, no 64-bit multiplies); only the value select is predicated.
  auto stage = [&](int k0, int buf) {
    if (a_cs == 1) {            // k contiguous: lanes sweep k
      float ar[8];
#pragma unroll
      for (int e = 0; e < 8; ++e) {
        int idx = tid + 256 * e;
        int r = idx >> 5, c = idx & 31;
        int m = m0 + r, k = k0 + c;
        bool p = (m < M) && (k < K);
        float v = A[imin(m, M - 1) * a_rs + imin(k, K - 1)];
        ar[e] = p ? v : 0.f;
      }
#pragma unroll
      for (int e = 0; e < 8; ++e) Asl[buf][tid + 256 * e] = (_Float16)ar[e];
    } else {                    // m contiguous: lanes sweep m
      float ar[8];
#pragma unroll
      for (int e = 0; e < 8; ++e) {
        int idx = e * 256 + tid;
        int r = idx & 63, c = idx >> 6;
        int m = m0 + r, k = k0 + c;
        bool p = (m < M) && (k < K);
        float v = A[imin(m, M - 1) * a_rs + imin(k, K - 1) * a_cs];
        ar[e] = p ? v : 0.f;
      }
#pragma unroll
      for (int e = 0; e < 8; ++e) {
        int idx = e * 256 + tid;
        Asl[buf][(idx & 63) * 32 + (idx >> 6)] = (_Float16)ar[e];
      }
    }
    if (b_rs == 1) {            // k contiguous: lanes sweep k
      float brg[16];
#pragma unroll
      for (int e = 0; e < 16; ++e) {
        int idx = tid + 256 * e;
        int r = idx >> 5, c = idx & 31;
        int n = n0 + r, k = k0 + c;
        bool p = (n < N) && (k < K);
        float v = B[imin(k, K - 1) + imin(n, N - 1) * b_cs];
        brg[e] = p ? v : 0.f;
      }
#pragma unroll
      for (int e = 0; e < 16; ++e) Bsl[buf][tid + 256 * e] = (_Float16)brg[e];
    } else {                    // n contiguous: lanes sweep n
      float brg[16];
#pragma unroll
      for (int e = 0; e < 16; ++e) {
        int idx = e * 256 + tid;
        int r = idx & 127, c = idx >> 7;
        int n = n0 + r, k = k0 + c;
        bool p = (n < N) && (k < K);
        float v = B[imin(k, K - 1) * b_rs + imin(n, N - 1) * b_cs];
        brg[e] = p ? v : 0.f;
      }
#pragma unroll
      for (int e = 0; e < 16; ++e) {
        int idx = e * 256 + tid;
        Bsl[buf][(idx & 127) * 32 + (idx >> 7)] = (_Float16)brg[e];
      }
    }
  };

  v8f acc[4] = {};
  const int nsteps = (K + 31) / 32;

  stage(0, 0);
  __syncthreads();
  for (int s = 0; s < nsteps; ++s) {
    const int cur = s & 1;
    if (s + 1 < nsteps) stage((s + 1) * 32, cur ^ 1);
    v16h a = load_a_frag(&Asl[cur][0], m_sub, lane);
#pragma unroll
    for (int j = 0; j < 4; ++j) {
      v16h bf = load_b_frag(&Bsl[cur][0], n_grp + j * 16, lane);
      acc[j] = WMMA_F16(a, bf, acc[j]);
    }
    __syncthreads();
  }

  const int rgrp = (lane >> 4) * 8;
  const int cn = lane & 15;
#pragma unroll
  for (int j = 0; j < 4; ++j) {
    int nb = n0 + n_grp + j * 16 + cn;
    if (nb < N) {
      float bv = bias ? bias[nb] : 0.f;
#pragma unroll
      for (int r = 0; r < 8; ++r) {
        int mg = m0 + m_sub + r + rgrp;
        if (mg < M) {
          float v = (acc[j][r] + bv) * alpha;
          if (relu) v = fmaxf(v, 0.f);
          D[mg * d_rs + nb] = v;
        }
      }
    }
  }
}

// ---------------------------------------------------------------------------
// Implicit-GEMM 3x3 SAME conv, NCHW/OIHW.  M = Cout, N = H*W, K = 9*Cin
// (k = cin*9 + tap so the OIHW weight read is contiguous in k).
// Weights staged with float4 loads + one ds_store_b128 per thread.
// Input gathered pixel-fastest: lanes sweep consecutive pixels at wave-uniform
// k -> contiguous global reads.  All offsets 32-bit.
// ---------------------------------------------------------------------------
__global__ __launch_bounds__(256) void k_conv3x3(
    const float* __restrict__ X, const float* __restrict__ Wt,
    const float* __restrict__ bias, float* __restrict__ Y,
    int Cin, int H, int Wd, int Cout, int relu)
{
  __shared__ __align__(16) _Float16 Asl[2][64 * 32];    // weights [cout][k]
  __shared__ __align__(16) _Float16 Bsl[2][128 * 32];   // input   Bt[pix][k]

  const int HW = H * Wd;
  const int K = Cin * 9;                 // 2304: multiple of 32
  const int b = blockIdx.z;
  const float* Xb = X + (long)b * Cin * HW;
  float* Yb = Y + (long)b * Cout * HW;
  const int m0 = blockIdx.x * 64;        // cout tile (Cout=256: always full)
  const int n0 = blockIdx.y * 128;       // pixel tile
  const int tid = threadIdx.x;
  const int lane = tid & 31;
  const int w = tid >> 5;
  const int m_sub = (w & 3) * 16;
  const int n_grp = (w >> 2) * 64;

  auto stage = [&](int k0, int buf) {
    // A: 8 contiguous weights per thread -> 2x float4 load, 1x ds_store_b128
    {
      int r = tid >> 2;                  // 0..63  (cout row)
      int c0 = (tid & 3) * 8;            // 0,8,16,24
      const float4* p = (const float4*)(Wt + (m0 + r) * K + k0 + c0);
      float4 w0 = p[0], w1 = p[1];
      v8h h;
      h[0] = (_Float16)w0.x; h[1] = (_Float16)w0.y;
      h[2] = (_Float16)w0.z; h[3] = (_Float16)w0.w;
      h[4] = (_Float16)w1.x; h[5] = (_Float16)w1.y;
      h[6] = (_Float16)w1.z; h[7] = (_Float16)w1.w;
      *(v8h*)(&Asl[buf][r * 32 + c0]) = h;
    }
    // B: pixel-fastest gather with clamped 32-bit addressing (branchless)
    float brg[16];
#pragma unroll
    for (int e = 0; e < 16; ++e) {
      int idx = e * 256 + tid;
      int r = idx & 127;                 // pixel (lane-fast)
      int c = idx >> 7;                  // k (wave-uniform)
      int n = n0 + r, k = k0 + c;
      int nc = imin(n, HW - 1);
      int cin = k / 9, tap = k % 9;
      int y = nc / Wd + tap / 3 - 1;
      int x = nc % Wd + tap % 3 - 1;
      bool p = (n < HW) && (y >= 0) && (y < H) && (x >= 0) && (x < Wd);
      int yc = imin(imax(y, 0), H - 1);
      int xc = imin(imax(x, 0), Wd - 1);
      float v = Xb[cin * HW + yc * Wd + xc];
      brg[e] = p ? v : 0.f;
    }
#pragma unroll
    for (int e = 0; e < 16; ++e) {
      int idx = e * 256 + tid;
      Bsl[buf][(idx & 127) * 32 + (idx >> 7)] = (_Float16)brg[e];
    }
  };

  v8f acc[4] = {};
  const int nsteps = K / 32;             // K = 2304 -> 72

  stage(0, 0);
  __syncthreads();
  for (int s = 0; s < nsteps; ++s) {
    const int cur = s & 1;
    if (s + 1 < nsteps) {
      // prefetch upcoming weight tile (lowers to global_prefetch_b8)
      __builtin_prefetch(Wt + (m0 + (tid >> 2)) * K + (s + 1) * 32, 0, 1);
      stage((s + 1) * 32, cur ^ 1);
    }
    v16h a = load_a_frag(&Asl[cur][0], m_sub, lane);
#pragma unroll
    for (int j = 0; j < 4; ++j) {
      v16h bf = load_b_frag(&Bsl[cur][0], n_grp + j * 16, lane);
      acc[j] = WMMA_F16(a, bf, acc[j]);
    }
    __syncthreads();
  }

  const int rgrp = (lane >> 4) * 8;
  const int cn = lane & 15;
#pragma unroll
  for (int j = 0; j < 4; ++j) {
    int nb = n0 + n_grp + j * 16 + cn;   // pixel -> consecutive lanes coalesce
    if (nb < HW) {
#pragma unroll
      for (int r = 0; r < 8; ++r) {
        int mg = m0 + m_sub + r + rgrp;  // cout (Cout=256 tiles exactly)
        float v = acc[j][r] + bias[mg];
        if (relu) v = fmaxf(v, 0.f);
        Yb[mg * HW + nb] = v;
      }
    }
  }
}

// ---------------------------------------------------------------------------
// GroupNorm(1): per-batch mean / rsqrt(var+eps) over C*H*W, then apply in place.
// ---------------------------------------------------------------------------
__global__ __launch_bounds__(256) void k_gn_stats(const float* __restrict__ X,
                                                  float* __restrict__ stats, long n) {
  const int b = blockIdx.x;
  const float* p = X + (long)b * n;
  float s = 0.f, ss = 0.f;
  for (long i = threadIdx.x; i < n; i += 256) { float v = p[i]; s += v; ss += v * v; }
  __shared__ float rs[256], rq[256];
  rs[threadIdx.x] = s; rq[threadIdx.x] = ss;
  __syncthreads();
  for (int o = 128; o > 0; o >>= 1) {
    if (threadIdx.x < o) { rs[threadIdx.x] += rs[threadIdx.x + o]; rq[threadIdx.x] += rq[threadIdx.x + o]; }
    __syncthreads();
  }
  if (threadIdx.x == 0) {
    float m = rs[0] / (float)n;
    float var = rq[0] / (float)n - m * m;
    stats[b * 2] = m;
    stats[b * 2 + 1] = rsqrtf(var + 1e-5f);
  }
}

__global__ __launch_bounds__(256) void k_gn_apply(float* __restrict__ X,
                                                  const float* __restrict__ stats,
                                                  long n, int relu) {
  const int b = blockIdx.y;
  const float m = stats[b * 2], r = stats[b * 2 + 1];
  float* p = X + (long)b * n;
  long i = (long)blockIdx.x * 256 + threadIdx.x;
  if (i < n) {
    float v = (p[i] - m) * r;
    if (relu) v = fmaxf(v, 0.f);
    p[i] = v;
  }
}

// LayerNorm(no affine) + ReLU over rows of width 256.
__global__ __launch_bounds__(256) void k_ln_relu(const float* __restrict__ X,
                                                 float* __restrict__ Y, int C) {
  const int row = blockIdx.x;
  float v = X[row * C + threadIdx.x];
  __shared__ float rs[256], rq[256];
  rs[threadIdx.x] = v; rq[threadIdx.x] = v * v;
  __syncthreads();
  for (int o = 128; o > 0; o >>= 1) {
    if (threadIdx.x < o) { rs[threadIdx.x] += rs[threadIdx.x + o]; rq[threadIdx.x] += rq[threadIdx.x + o]; }
    __syncthreads();
  }
  float m = rs[0] / (float)C;
  float var = rq[0] / (float)C - m * m;
  float y = (v - m) * rsqrtf(var + 1e-5f);
  Y[row * C + threadIdx.x] = fmaxf(y, 0.f);
}

// Row softmax, width 256.
__global__ __launch_bounds__(256) void k_softmax(float* __restrict__ S, int W) {
  const int row = blockIdx.x;
  float* p = S + (long)row * W;
  float v = p[threadIdx.x];
  __shared__ float red[256];
  red[threadIdx.x] = v;
  __syncthreads();
  for (int o = 128; o > 0; o >>= 1) {
    if (threadIdx.x < o) red[threadIdx.x] = fmaxf(red[threadIdx.x], red[threadIdx.x + o]);
    __syncthreads();
  }
  float mx = red[0];
  __syncthreads();
  float e = expf(v - mx);
  red[threadIdx.x] = e;
  __syncthreads();
  for (int o = 128; o > 0; o >>= 1) {
    if (threadIdx.x < o) red[threadIdx.x] += red[threadIdx.x + o];
    __syncthreads();
  }
  p[threadIdx.x] = e / red[0];
}

// stu[b,n,:] /= max(sum_p mask[b,n,p], 1)   (pool denominator epilogue)
__global__ __launch_bounds__(256) void k_rowdiv(float* __restrict__ stu,
                                                const float* __restrict__ mask,
                                                int Np, int C, int P) {
  const int b = blockIdx.y, n = blockIdx.x;
  const float* mp = mask + (b * Np + n) * P;
  float s = 0.f;
  for (int p = threadIdx.x; p < P; p += 256) s += mp[p];
  __shared__ float red[256];
  red[threadIdx.x] = s;
  __syncthreads();
  for (int o = 128; o > 0; o >>= 1) {
    if (threadIdx.x < o) red[threadIdx.x] += red[threadIdx.x + o];
    __syncthreads();
  }
  float denom = fmaxf(red[0], 1.f);
  float* row = stu + (b * Np + n) * C;
  row[threadIdx.x] = row[threadIdx.x] / denom;
}

// ---------------------------------------------------------------------------
extern "C" void kernel_launch(void* const* d_in, const int* in_sizes, int n_in,
                              void* d_out, int out_size, void* d_ws, size_t ws_size,
                              hipStream_t stream) {
  (void)in_sizes; (void)n_in; (void)out_size; (void)ws_size;
  const int C = 256, Bn = 8, Np = 32, T = Bn * Np;   // T = 256
  static const int HWS[5][2] = {{100,152},{50,76},{25,38},{13,19},{7,10}};

  const float* canoni = (const float*)d_in[0];
  const float* W_can = (const float*)d_in[11]; const float* b_can = (const float*)d_in[12];
  const float* W_stu = (const float*)d_in[13]; const float* b_stu = (const float*)d_in[14];
  const float* Wqkv  = (const float*)d_in[15]; const float* bqkv  = (const float*)d_in[16];
  const float* Wo    = (const float*)d_in[17]; const float* bo    = (const float*)d_in[18];
  const float* W_i1  = (const float*)d_in[19]; const float* b_i1  = (const float*)d_in[20];
  const float* W_i2  = (const float*)d_in[21]; const float* b_i2  = (const float*)d_in[22];
  const float* Wr[3] = {(const float*)d_in[23], (const float*)d_in[25], (const float*)d_in[27]};
  const float* br[3] = {(const float*)d_in[24], (const float*)d_in[26], (const float*)d_in[28]};

  // workspace carve (256B aligned)
  char* ws = (char*)d_ws;
  size_t off = 0;
  auto carve = [&](size_t bytes) -> float* {
    float* p = (float*)(ws + off);
    off = (off + bytes + 255) & ~(size_t)255;
    return p;
  };
  const long maxHW = 100L * 152;
  const size_t bigBytes = (size_t)Bn * C * maxHW * sizeof(float);
  float* big0  = carve(bigBytes);
  float* big1  = carve(bigBytes);
  float* tmp   = carve((size_t)T * C * sizeof(float));
  float* ce    = carve((size_t)T * C * sizeof(float));
  float* stu   = carve((size_t)T * C * sizeof(float));
  float* Qb    = carve((size_t)T * C * sizeof(float));
  float* Kb    = carve((size_t)T * C * sizeof(float));
  float* Vb    = carve((size_t)T * C * sizeof(float));
  float* Sb    = carve((size_t)8 * T * T * sizeof(float));
  float* Ob    = carve((size_t)T * C * sizeof(float));
  float* attnb = carve((size_t)T * C * sizeof(float));
  float* instb = carve((size_t)T * C * sizeof(float));
  float* stats = carve((size_t)Bn * 2 * sizeof(float));

  const dim3 blk(256);
  const dim3 g42(4, 2, 1);   // 256x256 GEMM: 4 M-tiles x 2 N-tiles

  // canoni_proj_1D: GEMM -> LN -> ReLU
  k_gemm<<<g42, blk, 0, stream>>>(canoni, W_can, b_can, tmp, T, C, C,
                                  C, 1, 1, C, C, 0, 0, 0, 1.0f, 0);
  k_ln_relu<<<dim3(T), blk, 0, stream>>>(tmp, ce, C);

  float* outp = (float*)d_out;
  for (int lvl = 0; lvl < 5; ++lvl) {
    const int H = HWS[lvl][0], Wd = HWS[lvl][1], HW = H * Wd, P = HW;
    const float* feat = (const float*)d_in[1 + 2 * lvl];
    const float* mask = (const float*)d_in[2 + 2 * lvl];
    const long nper = (long)C * HW;
    const dim3 gconv(4, (HW + 127) / 128, Bn);
    const dim3 ggn((unsigned)((nper + 255) / 256), Bn);

    // student_proj_2D: conv -> GN -> ReLU  (big0)
    k_conv3x3<<<gconv, blk, 0, stream>>>(feat, W_stu, b_stu, big0, C, H, Wd, C, 0);
    k_gn_stats<<<dim3(Bn), blk, 0, stream>>>(big0, stats, nper);
    k_gn_apply<<<ggn, blk, 0, stream>>>(big0, stats, nper, 1);

    // mask-weighted pooling as batched GEMM (M=32, N=256, K=P), then row-divide
    k_gemm<<<dim3(1, 2, Bn), blk, 0, stream>>>(mask, big0, nullptr, stu,
                                               Np, C, P,
                                               P, 1, 1, P, C,
                                               (long)Np * P, (long)C * P, (long)Np * C,
                                               1.0f, 0);
    k_rowdiv<<<dim3(Np, Bn), blk, 0, stream>>>(stu, mask, Np, C, P);

    // MHA: q from stu (scaled), k/v from ce
    const float scale = 1.0f / sqrtf(32.0f);
    k_gemm<<<g42, blk, 0, stream>>>(stu, Wqkv, bqkv, Qb, T, C, C,
                                    C, 1, 1, C, C, 0, 0, 0, scale, 0);
    k_gemm<<<g42, blk, 0, stream>>>(ce, Wqkv + C * C, bqkv + C, Kb, T, C, C,
                                    C, 1, 1, C, C, 0, 0, 0, 1.0f, 0);
    k_gemm<<<g42, blk, 0, stream>>>(ce, Wqkv + 2 * C * C, bqkv + 2 * C, Vb, T, C, C,
                                    C, 1, 1, C, C, 0, 0, 0, 1.0f, 0);
    // per-head scores S[h,q,t] = Qh @ Kh^T  (M=256, N=256, K=32; batch=8 heads)
    k_gemm<<<dim3(4, 2, 8), blk, 0, stream>>>(Qb, Kb, nullptr, Sb, T, T, 32,
                                              C, 1, 1, C, T, 32, 32, (long)T * T, 1.0f, 0);
    k_softmax<<<dim3(8 * T), blk, 0, stream>>>(Sb, T);
    // context O[q, h*32+d] = S[h] @ Vh  (M=256, N=32, K=256; batch=8 heads)
    k_gemm<<<dim3(4, 1, 8), blk, 0, stream>>>(Sb, Vb, nullptr, Ob, T, 32, T,
                                              T, 1, C, 1, C, (long)T * T, 32, 32, 1.0f, 0);
    // output proj, then instance proj
    k_gemm<<<g42, blk, 0, stream>>>(Ob, Wo, bo, attnb, T, C, C,
                                    C, 1, 1, C, C, 0, 0, 0, 1.0f, 0);
    k_gemm<<<g42, blk, 0, stream>>>(attnb, W_i1, b_i1, instb, T, C, C,
                                    C, 1, 1, C, C, 0, 0, 0, 1.0f, 0);

    // scatter-back warp as batched GEMM: warped[b,c,p] = sum_n inst[b,n,c]*m[b,n,p]
    // (M=C, N=P, K=32)
    k_gemm<<<dim3(4, (P + 127) / 128, Bn), blk, 0, stream>>>(
        instb, mask, nullptr, big1,
        C, P, Np,
        1, C, P, 1, P,
        (long)Np * C, (long)Np * P, (long)C * P,
        1.0f, 0);

    // local_inst_proj conv + ReLU (big0)
    k_conv3x3<<<gconv, blk, 0, stream>>>(big1, W_i2, b_i2, big0, C, H, Wd, C, 1);

    // refinement: (conv -> GN -> ReLU) x2, then conv -> GN (into d_out)
    k_conv3x3<<<gconv, blk, 0, stream>>>(big0, Wr[0], br[0], big1, C, H, Wd, C, 0);
    k_gn_stats<<<dim3(Bn), blk, 0, stream>>>(big1, stats, nper);
    k_gn_apply<<<ggn, blk, 0, stream>>>(big1, stats, nper, 1);

    k_conv3x3<<<gconv, blk, 0, stream>>>(big1, Wr[1], br[1], big0, C, H, Wd, C, 0);
    k_gn_stats<<<dim3(Bn), blk, 0, stream>>>(big0, stats, nper);
    k_gn_apply<<<ggn, blk, 0, stream>>>(big0, stats, nper, 1);

    k_conv3x3<<<gconv, blk, 0, stream>>>(big0, Wr[2], br[2], outp, C, H, Wd, C, 0);
    k_gn_stats<<<dim3(Bn), blk, 0, stream>>>(outp, stats, nper);
    k_gn_apply<<<ggn, blk, 0, stream>>>(outp, stats, nper, 0);

    outp += (long)Bn * C * HW;
  }
}
